// TSMixerH_28930899706365
// MI455X (gfx1250) — matrix operand order repaired and checked by reading, not compiled, not run on hardware
//
#include <hip/hip_runtime.h>

// Problem constants (match reference)
#define BB   32
#define LL   512
#define PP   96
#define CC   512
#define KCL  8
#define NLAY 2
#define DFF  512
#define CKN  64
#define EPSF 1e-5f

typedef __attribute__((ext_vector_type(16))) __bf16 v16bf;
typedef __attribute__((ext_vector_type(8)))  __bf16 v8bf;
typedef __attribute__((ext_vector_type(8)))  float  v8f;

// D = A(16x32 bf16) x B(32x16 bf16) + C(16x16 f32)   -> v_wmma_f32_16x16x32_bf16
__device__ __forceinline__ v8f wmma_bf16(v16bf a, v16bf b, v8f c) {
  return __builtin_amdgcn_wmma_f32_16x16x32_bf16(false, a, false, b, (short)0, c, false, false);
}

// 16 contiguous bf16 (B-operand: K = h + 16*(lane>>4) contiguous per lane)
__device__ __forceinline__ v16bf ld16(const __bf16* p) {
  v8bf lo = *(const v8bf*)p;
  v8bf hi = *(const v8bf*)(p + 8);
  v16bf r;
#pragma unroll
  for (int i = 0; i < 8; ++i) { r[i] = lo[i]; r[i + 8] = hi[i]; }
  return r;
}
// two groups of 8 bf16 (A-operand: K = h + 16*(lane>>4), contiguous as for B)
__device__ __forceinline__ v16bf ld2x8(const __bf16* p0, const __bf16* p1) {
  v8bf lo = *(const v8bf*)p0;
  v8bf hi = *(const v8bf*)p1;
  v16bf r;
#pragma unroll
  for (int i = 0; i < 8; ++i) { r[i] = lo[i]; r[i + 8] = hi[i]; }
  return r;
}

// ---------------- setup: stable-argsort(cluster_ids % K) -> idxmap[K][CK] -----
__global__ void setup_perm_kernel(const int* __restrict__ cluster_ids,
                                  int* __restrict__ idxmap) {
  if (threadIdx.x == 0 && blockIdx.x == 0) {
    int count[KCL];
    for (int k = 0; k < KCL; ++k) count[k] = 0;
    for (int c = 0; c < CC; ++c) {
      int a = cluster_ids[c] % KCL; if (a < 0) a += KCL;
      if (count[a] < CKN) { idxmap[a * CKN + count[a]] = c; ++count[a]; }
    }
  }
}

// ------------- tiled transpose + fp32 -> bf16 convert: out[b][c][r] = in[b][r][c]
__global__ void transpose_cvt_kernel(const float* __restrict__ in,
                                     __bf16* __restrict__ out,
                                     int rows, int cols) {
  __shared__ float tile[16][17];
  const int bz = blockIdx.z;
  const int r0 = blockIdx.y * 16, c0 = blockIdx.x * 16;
  const int tx = threadIdx.x, ty = threadIdx.y;
  const float* ip = in + (size_t)bz * rows * cols;
  __bf16* op = out + (size_t)bz * rows * cols;
  tile[ty][tx] = ip[(size_t)(r0 + ty) * cols + (c0 + tx)];
  __syncthreads();
  op[(size_t)(c0 + ty) * rows + (r0 + tx)] = (__bf16)tile[tx][ty];
}

// ---------------- LayerNorm over channel dim, write bf16 (opt. transposed) -----
__device__ __forceinline__ void ln_to_bf16(float* Xs, float* rowM, float* rowR,
                                           const float* lng, const float* lnb,
                                           __bf16* dst, bool transposed, int tid) {
  for (int l = tid; l < LL; l += 256) {
    float s = 0.f, s2 = 0.f;
    for (int c = 0; c < CKN; ++c) { float v = Xs[l * 65 + c]; s += v; s2 += v * v; }
    float m = s * (1.0f / CKN);
    float var = s2 * (1.0f / CKN) - m * m;
    rowM[l] = m; rowR[l] = rsqrtf(var + EPSF);
  }
  __syncthreads();
  for (int idx = tid; idx < LL * CKN; idx += 256) {
    int l = idx >> 6, c = idx & 63;
    float y = (Xs[l * 65 + c] - rowM[l]) * rowR[l] * lng[c] + lnb[c];
    if (transposed) dst[c * 520 + l] = (__bf16)y;   // Yt[c][l]
    else            dst[l * 72 + c]  = (__bf16)y;   // Yn[l][c]
  }
  __syncthreads();
}

// ---------------------------- fused mixer kernel ------------------------------
// one workgroup per (k, b); 256 threads = 8 waves (wave32)
__global__ __launch_bounds__(256, 1)
void mixer_main_kernel(const float* __restrict__ x,
                       const float* __restrict__ rev_w,
                       const float* __restrict__ rev_b,
                       const float* __restrict__ bt,
                       const float* __restrict__ ln1_g, const float* __restrict__ ln1_b,
                       const float* __restrict__ bf1,
                       const float* __restrict__ bf2,
                       const float* __restrict__ ln2_g, const float* __restrict__ ln2_b,
                       const float* __restrict__ bout,
                       const __bf16* __restrict__ WtT,    // [k,l][m][l'] bf16
                       const __bf16* __restrict__ Wf1T,   // [k,l][f][c]  bf16
                       const __bf16* __restrict__ Wf2T,   // [k,l][c][f]  bf16
                       const __bf16* __restrict__ WoutT,  // [k][p][l]    bf16
                       const int* __restrict__ idxmap,
                       float* __restrict__ out) {
  extern __shared__ char smem[];
  float*  Xs    = (float*)smem;                          // [512][65] f32 residual state
  __bf16* Ybuf  = (__bf16*)(smem + 133120);              // union Yt[64][520] / Yn[512][72]
  __bf16* Hs    = (__bf16*)(smem + 133120 + 73728);      // [8 waves][16][56]
  char* pm = smem + 133120 + 73728 + 14336;
  float* rowM  = (float*)pm; pm += 2048;
  float* rowR  = (float*)pm; pm += 2048;
  float* chanA = (float*)pm; pm += 256;
  float* chanB = (float*)pm; pm += 256;
  float* chanC = (float*)pm; pm += 256;
  float* chanD = (float*)pm; pm += 256;
  float* lng   = (float*)pm; pm += 256;
  float* lnb   = (float*)pm; pm += 256;
  float* btS   = (float*)pm; pm += 2048;
  float* bf1S  = (float*)pm; pm += 2048;
  float* bf2S  = (float*)pm; pm += 256;
  float* red   = (float*)pm; pm += 1024;
  float* red2  = (float*)pm; pm += 1024;
  int*   chanIdx = (int*)pm;

  const int tid  = threadIdx.x;
  const int b    = blockIdx.x & (BB - 1);
  const int k    = blockIdx.x >> 5;
  const int wv   = tid >> 5;
  const int lane = tid & 31;
  const int ln15 = lane & 15;
  const int lhi  = lane >> 4;

  if (tid < CKN) chanIdx[tid] = idxmap[k * CKN + tid];
  __syncthreads();

  // ---- phase 0: gather cluster channels + RevIN (L2 absorbs the strided reads)
  for (int idx = tid; idx < LL * CKN; idx += 256) {
    int l = idx >> 6, j = idx & 63;
    Xs[l * 65 + j] = x[((size_t)b * LL + l) * CC + chanIdx[j]];
  }
  __syncthreads();
  {
    int j = tid & 63, part = tid >> 6;
    float s = 0.f, s2 = 0.f;
    for (int l = part * 128; l < part * 128 + 128; ++l) {
      float v = Xs[l * 65 + j]; s += v; s2 += v * v;
    }
    red[part * 64 + j] = s; red2[part * 64 + j] = s2;
  }
  __syncthreads();
  if (tid < CKN) {
    int j = tid;
    float s  = red[j]  + red[64 + j]  + red[128 + j]  + red[192 + j];
    float s2 = red2[j] + red2[64 + j] + red2[128 + j] + red2[192 + j];
    float mean = s * (1.f / LL);
    float var  = s2 * (1.f / LL) - mean * mean;
    float stdv = sqrtf(var + EPSF);
    int cg = chanIdx[j];
    float rw = rev_w[cg], rb = rev_b[cg];
    float a = rw / stdv;
    chanA[j] = a;                   // norm scale
    chanB[j] = rb - mean * a;       // norm bias
    float cc = stdv / rw;
    chanC[j] = cc;                  // denorm scale
    chanD[j] = mean - rb * cc;      // denorm bias
  }
  __syncthreads();
  for (int idx = tid; idx < LL * CKN; idx += 256) {
    int l = idx >> 6, j = idx & 63;
    Xs[l * 65 + j] = Xs[l * 65 + j] * chanA[j] + chanB[j];
  }
  __syncthreads();

  // ---- mixer layers ----
  for (int layer = 0; layer < NLAY; ++layer) {
    const int kl = k * NLAY + layer;

    // stage LN1 params + time-mix bias
    if (tid < CKN) { lng[tid] = ln1_g[kl * CKN + tid]; lnb[tid] = ln1_b[kl * CKN + tid]; }
    for (int m = tid; m < LL; m += 256) btS[m] = bt[kl * LL + m];
    ln_to_bf16(Xs, rowM, rowR, lng, lnb, Ybuf, /*transposed=*/true, tid);

    // ---- time mixing: X[m,c] += relu( sum_l Wt[l,m]*LN1(X)[l,c] + bt[m] )
    {
      const __bf16* WtTk = WtT + (size_t)kl * LL * LL;     // [m][l]
      for (int mt = wv; mt < 32; mt += 8) {
        v8f acc[4] = {};
        const __bf16* arow = WtTk + (size_t)(mt * 16 + ln15) * LL;
        for (int l0 = 0; l0 < LL; l0 += 32) {
          v16bf av = ld2x8(arow + l0 + 16 * lhi, arow + l0 + 8 + 16 * lhi);
#pragma unroll
          for (int ct = 0; ct < 4; ++ct) {
            v16bf bv = ld16(&Ybuf[(ct * 16 + ln15) * 520 + l0 + 16 * lhi]);
            acc[ct] = wmma_bf16(av, bv, acc[ct]);
          }
        }
#pragma unroll
        for (int r = 0; r < 8; ++r) {
          int m = mt * 16 + r + 8 * lhi;
          float bias = btS[m];
#pragma unroll
          for (int ct = 0; ct < 4; ++ct) {
            int c = ct * 16 + ln15;
            float v = acc[ct][r] + bias;
            v = v > 0.f ? v : 0.f;
            Xs[m * 65 + c] += v;
          }
        }
      }
    }
    __syncthreads();

    // stage LN2 params + MLP biases
    if (tid < CKN) {
      lng[tid] = ln2_g[kl * CKN + tid]; lnb[tid] = ln2_b[kl * CKN + tid];
      bf2S[tid] = bf2[kl * CKN + tid];
    }
    for (int f = tid; f < DFF; f += 256) bf1S[f] = bf1[kl * DFF + f];
    ln_to_bf16(Xs, rowM, rowR, lng, lnb, Ybuf, /*transposed=*/false, tid);

    // ---- feature mixing, fused: X += relu(LN2(X)*Wf1 + bf1) * Wf2 + bf2
    {
      const __bf16* Wf1Tk = Wf1T + (size_t)kl * DFF * CKN;  // [f][c] stride 64
      const __bf16* Wf2Tk = Wf2T + (size_t)kl * DFF * CKN;  // [c][f] stride 512
      __bf16* HsW = Hs + wv * 16 * 56;                      // per-wave staging
      for (int lt = wv; lt < 32; lt += 8) {
        v8f acc2[4] = {};
        const __bf16* aYrow = Ybuf + (size_t)(lt * 16 + ln15) * 72;
        for (int f0 = 0; f0 < DFF; f0 += 32) {
          v8f h0 = {}, h1 = {};
#pragma unroll
          for (int c0 = 0; c0 < CKN; c0 += 32) {
            v16bf av = ld2x8(aYrow + c0 + 16 * lhi, aYrow + c0 + 8 + 16 * lhi);
            v16bf b0 = ld16(Wf1Tk + (size_t)(f0 + ln15) * CKN + c0 + 16 * lhi);
            v16bf b1 = ld16(Wf1Tk + (size_t)(f0 + 16 + ln15) * CKN + c0 + 16 * lhi);
            h0 = wmma_bf16(av, b0, h0);
            h1 = wmma_bf16(av, b1, h1);
          }
          float bv0 = bf1S[f0 + ln15];
          float bv1 = bf1S[f0 + 16 + ln15];
#pragma unroll
          for (int r = 0; r < 8; ++r) {       // D layout -> Hs[l'][f'] (A layout src)
            int lp = r + 8 * lhi;
            float u0 = h0[r] + bv0; u0 = u0 > 0.f ? u0 : 0.f;
            float u1 = h1[r] + bv1; u1 = u1 > 0.f ? u1 : 0.f;
            HsW[lp * 56 + ln15]      = (__bf16)u0;
            HsW[lp * 56 + 16 + ln15] = (__bf16)u1;
          }
          __builtin_amdgcn_wave_barrier();
          asm volatile("s_wait_dscnt 0x0" ::: "memory");    // cross-lane LDS handoff
          v16bf a2 = ld2x8(HsW + ln15 * 56 + 16 * lhi, HsW + ln15 * 56 + 8 + 16 * lhi);
#pragma unroll
          for (int ct = 0; ct < 4; ++ct) {
            v16bf b2 = ld16(Wf2Tk + (size_t)(ct * 16 + ln15) * DFF + f0 + 16 * lhi);
            acc2[ct] = wmma_bf16(a2, b2, acc2[ct]);
          }
          __builtin_amdgcn_wave_barrier();
        }
#pragma unroll
        for (int r = 0; r < 8; ++r) {
          int l = lt * 16 + r + 8 * lhi;
#pragma unroll
          for (int ct = 0; ct < 4; ++ct) {
            int c = ct * 16 + ln15;
            Xs[l * 65 + c] += acc2[ct][r] + bf2S[c];
          }
        }
      }
    }
    __syncthreads();
  }

  // ---- output projection L -> P, + bout, RevIN denorm, scatter to d_out ----
  for (int idx = tid; idx < LL * CKN; idx += 256) {         // Xt[c][l] bf16 (B operand)
    int l = idx >> 6, c = idx & 63;
    Ybuf[c * 520 + l] = (__bf16)Xs[l * 65 + c];
  }
  __syncthreads();
  {
    const __bf16* WoTk = WoutT + (size_t)k * PP * LL;       // [p][l]
    for (int tt = wv; tt < 24; tt += 8) {                   // 6 p-tiles x 4 c-tiles
      int pt = tt >> 2, ct = tt & 3;
      v8f acc = {};
      const __bf16* arow = WoTk + (size_t)(pt * 16 + ln15) * LL;
      for (int l0 = 0; l0 < LL; l0 += 32) {
        v16bf av = ld2x8(arow + l0 + 16 * lhi, arow + l0 + 8 + 16 * lhi);
        v16bf bv = ld16(&Ybuf[(ct * 16 + ln15) * 520 + l0 + 16 * lhi]);
        acc = wmma_bf16(av, bv, acc);
      }
      int j = ct * 16 + ln15;
      int cg = chanIdx[j];
      float cCv = chanC[j], cDv = chanD[j];
#pragma unroll
      for (int r = 0; r < 8; ++r) {
        int p = pt * 16 + r + 8 * lhi;
        float v = acc[r] + bout[k * PP + p];
        out[((size_t)b * PP + p) * CC + cg] = v * cCv + cDv;
      }
    }
  }
}

// ------------------------------------------------------------------------------
extern "C" void kernel_launch(void* const* d_in, const int* in_sizes, int n_in,
                              void* d_out, int out_size, void* d_ws, size_t ws_size,
                              hipStream_t stream) {
  (void)in_sizes; (void)n_in; (void)out_size; (void)ws_size;
  const float* x           = (const float*)d_in[0];
  const int*   cluster_ids = (const int*)d_in[1];
  const float* rev_w       = (const float*)d_in[2];
  const float* rev_b       = (const float*)d_in[3];
  const float* Wt          = (const float*)d_in[4];
  const float* bt          = (const float*)d_in[5];
  const float* ln1_g       = (const float*)d_in[6];
  const float* ln1_b       = (const float*)d_in[7];
  const float* Wf1         = (const float*)d_in[8];
  const float* bf1         = (const float*)d_in[9];
  const float* Wf2         = (const float*)d_in[10];
  const float* bf2         = (const float*)d_in[11];
  const float* ln2_g       = (const float*)d_in[12];
  const float* ln2_b       = (const float*)d_in[13];
  const float* Wout        = (const float*)d_in[14];
  const float* bout        = (const float*)d_in[15];
  float* out = (float*)d_out;

  char* ws = (char*)d_ws;
  __bf16* WtT   = (__bf16*)ws; ws += (size_t)KCL * NLAY * LL * LL * 2;   // 8.0 MB
  __bf16* Wf1T  = (__bf16*)ws; ws += (size_t)KCL * NLAY * CKN * DFF * 2; // 1.0 MB
  __bf16* Wf2T  = (__bf16*)ws; ws += (size_t)KCL * NLAY * DFF * CKN * 2; // 1.0 MB
  __bf16* WoutT = (__bf16*)ws; ws += (size_t)KCL * PP * LL * 2;          // 0.75 MB
  int*    idxmap = (int*)ws;

  setup_perm_kernel<<<1, 1, 0, stream>>>(cluster_ids, idxmap);
  // weight transpose+convert: out[b][c][r] = (bf16) in[b][r][c]
  transpose_cvt_kernel<<<dim3(LL / 16, LL / 16, KCL * NLAY), dim3(16, 16), 0, stream>>>(
      Wt, WtT, LL, LL);                                   // -> WtT[m][l]
  transpose_cvt_kernel<<<dim3(DFF / 16, CKN / 16, KCL * NLAY), dim3(16, 16), 0, stream>>>(
      Wf1, Wf1T, CKN, DFF);                               // -> Wf1T[f][c]
  transpose_cvt_kernel<<<dim3(CKN / 16, DFF / 16, KCL * NLAY), dim3(16, 16), 0, stream>>>(
      Wf2, Wf2T, DFF, CKN);                               // -> Wf2T[c][f]
  transpose_cvt_kernel<<<dim3(PP / 16, LL / 16, KCL), dim3(16, 16), 0, stream>>>(
      Wout, WoutT, LL, PP);                               // -> WoutT[p][l]

  const size_t smem = 233472;  // 228 KB of the 320 KB WGP LDS
  mixer_main_kernel<<<dim3(KCL * BB), dim3(256), smem, stream>>>(
      x, rev_w, rev_b, bt, ln1_g, ln1_b, bf1, bf2, ln2_g, ln2_b, bout,
      WtT, Wf1T, Wf2T, WoutT, idxmap, out);
}
